// NonLocalBlock_72773925863745
// MI455X (gfx1250) — compile-verified, matrix operand order checked by reading, not run on hardware
//
#include <hip/hip_runtime.h>

typedef __attribute__((ext_vector_type(16))) _Float16 v16h;
typedef __attribute__((ext_vector_type(8)))  _Float16 v8h;
typedef __attribute__((ext_vector_type(8)))  float    v8f;

#define TM 128
#define TN 128
#define TK 32
#define LDS_STRIDE 40   // 32 + 8 halfs pad -> conflict-free b128 frag loads

// ---------------- f32 -> f16 conversion ----------------
__global__ __launch_bounds__(256) void f32_to_f16_kernel(const float* __restrict__ src,
                                                         _Float16* __restrict__ dst, int n) {
  int i = blockIdx.x * blockDim.x + threadIdx.x;
  int stride = gridDim.x * blockDim.x;
  for (; i < n; i += stride) dst[i] = (_Float16)src[i];
}

// ---------------- generic tiled WMMA GEMM (double-buffered LDS) ----------------
// C[M,N] = A[M,K] * B[K,N], A/B f16 row-major, f32 accumulate.
// Block tile 128x128, 8 wave32 waves as 4(M) x 2(N), wave tile 32x64 = 2x4 WMMA frags.
// MODE 0: outH[row*ldc+col] = h(acc)
// MODE 1: outH[row*ldc+col] = h(acc + extra[row*ldc+col])      (g + pos_decoder)
// MODE 2: outH[col*ldc+row] = h(acc)                           (transposed store: phi^T)
// MODE 3: outF[row*ldc+col] = acc + extra[col*ldc+row]         (S^T = phi^T theta + pos^T)
// MODE 4: outF[row*ldc+col] = acc + extra[row*ldc+col]         (final: + residual x)
template <int MODE>
__global__ __launch_bounds__(256) void wmma_gemm_kernel(
    const _Float16* __restrict__ A, int lda,
    const _Float16* __restrict__ B, int ldb,
    float* __restrict__ outF, _Float16* __restrict__ outH, int ldc,
    const float* __restrict__ extra, int K) {
  __shared__ _Float16 As[2][TM * LDS_STRIDE];
  __shared__ _Float16 Bs[2][TN * LDS_STRIDE];  // transposed: Bs[buf][n][k]

  const int t     = threadIdx.x;
  const int lane  = t & 31;
  const int wid   = t >> 5;     // 8 waves
  const int waveM = wid & 3;    // 4 waves along M (32 rows each)
  const int waveN = wid >> 2;   // 2 waves along N (64 cols each)
  const int mBase = blockIdx.y * TM;
  const int nBase = blockIdx.x * TN;

  const int r16  = lane & 15;
  const int hsel = lane >> 4;   // half-select per ISA 16-bit A/B fragment layout

  // per-thread global staging coordinates (constant across K-steps)
  // A tile 128x32: 512 chunks of 8 halfs, 2 per thread
  const int arow0 = t >> 2,          akc0 = (t & 3) << 3;
  const int arow1 = (t + 256) >> 2,  akc1 = ((t + 256) & 3) << 3;
  // B tile 32x128: 512 chunks of 8 halfs, 2 per thread
  const int bk0 = t >> 4,            bn0 = (t & 15) << 3;
  const int bk1 = (t + 256) >> 4,    bn1 = ((t + 256) & 15) << 3;

  const _Float16* Arow0 = A + (size_t)(mBase + arow0) * lda + akc0;
  const _Float16* Arow1 = A + (size_t)(mBase + arow1) * lda + akc1;
  const _Float16* Brow0 = B + (size_t)bk0 * ldb + nBase + bn0;
  const _Float16* Brow1 = B + (size_t)bk1 * ldb + nBase + bn1;

  v8f acc[2][4] = {};

  const int ns = K / TK;

  // ---- prologue: stage tile 0 into buffer 0 ----
  {
    v8h a0 = *(const v8h*)(Arow0);
    v8h a1 = *(const v8h*)(Arow1);
    v8h b0 = *(const v8h*)(Brow0);
    v8h b1 = *(const v8h*)(Brow1);
    *(v8h*)(&As[0][arow0 * LDS_STRIDE + akc0]) = a0;
    *(v8h*)(&As[0][arow1 * LDS_STRIDE + akc1]) = a1;
#pragma unroll
    for (int j = 0; j < 8; ++j) Bs[0][(bn0 + j) * LDS_STRIDE + bk0] = b0[j];
#pragma unroll
    for (int j = 0; j < 8; ++j) Bs[0][(bn1 + j) * LDS_STRIDE + bk1] = b1[j];
  }
  __syncthreads();

  for (int s = 0; s < ns; ++s) {
    const int cur = s & 1;
    const int nxt = cur ^ 1;
    const bool more = (s + 1) < ns;

    // ---- issue next tile's global loads early (overlap with WMMA) ----
    v8h a0r, a1r, b0r, b1r;
    if (more) {
      const int koff = (s + 1) * TK;
      a0r = *(const v8h*)(Arow0 + koff);
      a1r = *(const v8h*)(Arow1 + koff);
      b0r = *(const v8h*)(Brow0 + (size_t)koff * ldb);
      b1r = *(const v8h*)(Brow1 + (size_t)koff * ldb);
    }

    // ---- fragments from current buffer ----
    v16h afrag[2], bfrag[4];
#pragma unroll
    for (int fm = 0; fm < 2; ++fm) {
      int row = waveM * 32 + fm * 16 + r16;
      v8h lo = *(const v8h*)(&As[cur][row * LDS_STRIDE + hsel * 8]);       // K = hsel*8..+7
      v8h hi = *(const v8h*)(&As[cur][row * LDS_STRIDE + 16 + hsel * 8]);  // K = 16+hsel*8..+7
#pragma unroll
      for (int j = 0; j < 8; ++j) { afrag[fm][j] = lo[j]; afrag[fm][j + 8] = hi[j]; }
    }
#pragma unroll
    for (int fn = 0; fn < 4; ++fn) {
      int col = waveN * 64 + fn * 16 + r16;
      v8h lo = *(const v8h*)(&Bs[cur][col * LDS_STRIDE + hsel * 8]);
      v8h hi = *(const v8h*)(&Bs[cur][col * LDS_STRIDE + 16 + hsel * 8]);
#pragma unroll
      for (int j = 0; j < 8; ++j) { bfrag[fn][j] = lo[j]; bfrag[fn][j + 8] = hi[j]; }
    }

#pragma unroll
    for (int fm = 0; fm < 2; ++fm)
#pragma unroll
      for (int fn = 0; fn < 4; ++fn)
        acc[fm][fn] = __builtin_amdgcn_wmma_f32_16x16x32_f16(
            false, afrag[fm], false, bfrag[fn], (short)0, acc[fm][fn], false, false);

    // ---- store next tile into the other buffer; single barrier per step ----
    if (more) {
      *(v8h*)(&As[nxt][arow0 * LDS_STRIDE + akc0]) = a0r;
      *(v8h*)(&As[nxt][arow1 * LDS_STRIDE + akc1]) = a1r;
#pragma unroll
      for (int j = 0; j < 8; ++j) Bs[nxt][(bn0 + j) * LDS_STRIDE + bk0] = b0r[j];
#pragma unroll
      for (int j = 0; j < 8; ++j) Bs[nxt][(bn1 + j) * LDS_STRIDE + bk1] = b1r[j];
      __syncthreads();
    }
  }

  // ---- epilogue: C/D layout VGPR r -> M = r + hsel*8, lane%16 -> N ----
#pragma unroll
  for (int fm = 0; fm < 2; ++fm) {
#pragma unroll
    for (int fn = 0; fn < 4; ++fn) {
      int col = nBase + waveN * 64 + fn * 16 + r16;
#pragma unroll
      for (int r = 0; r < 8; ++r) {
        int row = mBase + waveM * 32 + fm * 16 + hsel * 8 + r;
        float v = acc[fm][fn][r];
        if (MODE == 0) {
          outH[(size_t)row * ldc + col] = (_Float16)v;
        } else if (MODE == 1) {
          outH[(size_t)row * ldc + col] = (_Float16)(v + extra[(size_t)row * ldc + col]);
        } else if (MODE == 2) {
          outH[(size_t)col * ldc + row] = (_Float16)v;
        } else if (MODE == 3) {
          outF[(size_t)row * ldc + col] = v + extra[(size_t)col * ldc + row];
        } else {
          outF[(size_t)row * ldc + col] = v + extra[(size_t)row * ldc + col];
        }
      }
    }
  }
}

// ---------------- row softmax: S^T rows -> P (f16) ----------------
__global__ __launch_bounds__(256) void softmax_rows_kernel(const float* __restrict__ S,
                                                           _Float16* __restrict__ P, int n) {
  __shared__ float red[256];
  const int row = blockIdx.x;
  const int t = threadIdx.x;
  const float* srow = S + (size_t)row * n;
  float vals[16];
  float m = -3.0e38f;
#pragma unroll
  for (int i = 0; i < 16; ++i) {
    vals[i] = srow[t + i * 256];
    m = fmaxf(m, vals[i]);
  }
  red[t] = m;
  __syncthreads();
  for (int s = 128; s > 0; s >>= 1) {
    if (t < s) red[t] = fmaxf(red[t], red[t + s]);
    __syncthreads();
  }
  m = red[0];
  __syncthreads();
  float sum = 0.f;
#pragma unroll
  for (int i = 0; i < 16; ++i) {
    vals[i] = __expf(vals[i] - m);
    sum += vals[i];
  }
  red[t] = sum;
  __syncthreads();
  for (int s = 128; s > 0; s >>= 1) {
    if (t < s) red[t] += red[t + s];
    __syncthreads();
  }
  float inv = 1.0f / red[0];
  _Float16* prow = P + (size_t)row * n;
#pragma unroll
  for (int i = 0; i < 16; ++i) prow[t + i * 256] = (_Float16)(vals[i] * inv);
}

// ---------------- host ----------------
extern "C" void kernel_launch(void* const* d_in, const int* in_sizes, int n_in,
                              void* d_out, int out_size, void* d_ws, size_t ws_size,
                              hipStream_t stream) {
  (void)in_sizes; (void)n_in; (void)out_size; (void)ws_size;

  const float* x       = (const float*)d_in[0];  // [8,512,64,64]
  const float* w_phi   = (const float*)d_in[1];  // [256,512]
  const float* w_theta = (const float*)d_in[2];  // [256,512]
  const float* w_g     = (const float*)d_in[3];  // [256,512]
  const float* w_mask  = (const float*)d_in[4];  // [512,256]
  const float* pos     = (const float*)d_in[5];  // [1,4096,4096]
  const float* posdec  = (const float*)d_in[6];  // [1,256,64,64]
  float* out = (float*)d_out;

  const int B = 8, C = 512, IC = 256, N = 4096;

  char* ws = (char*)d_ws;
  size_t off = 0;
  auto alloc = [&](size_t bytes) -> void* {
    void* p = ws + off;
    off = (off + bytes + 255) & ~(size_t)255;
    return p;
  };
  _Float16* xh      = (_Float16*)alloc((size_t)B * C * N * 2);
  _Float16* wphiH   = (_Float16*)alloc((size_t)IC * C * 2);
  _Float16* wthetaH = (_Float16*)alloc((size_t)IC * C * 2);
  _Float16* wgH     = (_Float16*)alloc((size_t)IC * C * 2);
  _Float16* wmaskH  = (_Float16*)alloc((size_t)C * IC * 2);
  _Float16* thetaH  = (_Float16*)alloc((size_t)IC * N * 2);  // [ic, n]
  _Float16* phiT    = (_Float16*)alloc((size_t)N * IC * 2);  // [n, ic]
  _Float16* gH      = (_Float16*)alloc((size_t)IC * N * 2);  // [ic, n]
  _Float16* YT      = (_Float16*)alloc((size_t)IC * N * 2);  // [ic, n]
  float*    Sb      = (float*)   alloc((size_t)N * N * 4);   // S^T [j, i]
  _Float16* P       = (_Float16*)alloc((size_t)N * N * 2);   // softmax'd S^T

  // one-time f32 -> f16 conversions
  f32_to_f16_kernel<<<2048, 256, 0, stream>>>(x, xh, B * C * N);
  f32_to_f16_kernel<<<128, 256, 0, stream>>>(w_phi, wphiH, IC * C);
  f32_to_f16_kernel<<<128, 256, 0, stream>>>(w_theta, wthetaH, IC * C);
  f32_to_f16_kernel<<<128, 256, 0, stream>>>(w_g, wgH, IC * C);
  f32_to_f16_kernel<<<128, 256, 0, stream>>>(w_mask, wmaskH, C * IC);

  dim3 blk(256);
  for (int b = 0; b < B; ++b) {
    const _Float16* xb = xh + (size_t)b * C * N;   // [C, N] f16
    // theta = Wtheta @ x        [IC, N]
    wmma_gemm_kernel<0><<<dim3(N / TN, IC / TM), blk, 0, stream>>>(
        wthetaH, C, xb, N, nullptr, thetaH, N, nullptr, C);
    // phi^T = (Wphi @ x)^T      [N, IC]
    wmma_gemm_kernel<2><<<dim3(N / TN, IC / TM), blk, 0, stream>>>(
        wphiH, C, xb, N, nullptr, phiT, IC, nullptr, C);
    // g = Wg @ x + pos_decoder  [IC, N]
    wmma_gemm_kernel<1><<<dim3(N / TN, IC / TM), blk, 0, stream>>>(
        wgH, C, xb, N, nullptr, gH, N, posdec, C);
    // S^T[j,i] = sum_c phi[c,j] theta[c,i] + pos[i,j]   [N, N]
    wmma_gemm_kernel<3><<<dim3(N / TN, N / TM), blk, 0, stream>>>(
        phiT, IC, thetaH, N, Sb, nullptr, N, pos, IC);
    // softmax over i-axis == rows of S^T
    softmax_rows_kernel<<<N, 256, 0, stream>>>(Sb, P, N);
    // Y^T[c,i] = sum_j g[c,j] P[j,i]     [IC, N]
    wmma_gemm_kernel<0><<<dim3(N / TN, IC / TM), blk, 0, stream>>>(
        gH, N, P, N, nullptr, YT, N, nullptr, N);
    // out = Wmask @ Y^T + x     [C, N]
    wmma_gemm_kernel<4><<<dim3(N / TN, C / TM), blk, 0, stream>>>(
        wmaskH, IC, YT, N, out + (size_t)b * C * N, nullptr, N,
        x + (size_t)b * C * N, IC);
  }
}
